// ScaledDotProductAttention_33964601376785
// MI455X (gfx1250) — matrix operand order repaired and checked by reading, not compiled, not run on hardware
//
#include <hip/hip_runtime.h>

// ---- vector types ----
typedef __attribute__((ext_vector_type(16))) _Float16 v16h;
typedef __attribute__((ext_vector_type(8)))  _Float16 v8h;
typedef __attribute__((ext_vector_type(4)))  _Float16 v4h;
typedef __attribute__((ext_vector_type(2)))  _Float16 v2h;
typedef __attribute__((ext_vector_type(8)))  float    v8f;
typedef __attribute__((ext_vector_type(4)))  float    v4f;

#define NSEQ   8192
#define DMODEL 1024
#define BM     32      // query rows per workgroup (two WMMA M tiles)
#define BN     32      // key/value rows per j iteration
#define NTHR   512     // 16 waves of 32
#define KSPAN  128     // per-wave slice of the head dimension

// Scheduler pinning (DS_READ=0x100, DS_WRITE=0x200, MFMA/WMMA=0x8)
#if defined(__has_builtin)
#  if __has_builtin(__builtin_amdgcn_sched_group_barrier)
#    define SGB(mask, n, id) __builtin_amdgcn_sched_group_barrier((mask), (n), (id))
#  endif
#endif
#ifndef SGB
#  define SGB(mask, n, id)
#endif

// LDS layout (bytes):
//   Klds : 32*1024 f16             = 65536
//   Vt   : 1024*32 f16 (transposed)= 65536
//   Sred : 2g*2nt*8r*32lane*8w f32 = 32768
//   Plds : 16 waves * 16*32 f16    = 16384
#define OFF_K  0
#define OFF_V  65536
#define OFF_S  131072
#define OFF_P  163840
#define SMEM_BYTES (163840 + 16384)   // 180224 B < 320 KB WGP LDS

static __device__ inline v8f wmma16(v16h a, v16h b, v8f c) {
  return __builtin_amdgcn_wmma_f32_16x16x32_f16(false, a, false, b, (short)0, c,
                                                false, false);
}

// A-fragment (16x32 f16) straight from global f32: per lane, 8 f16 at p[0..7]
// and 8 f16 at p[16..23] (caller bakes the lane-half +8 offset into p).
static __device__ inline v16h a_frag_f32(const float* p) {
  v4f a = *(const v4f*)(p + 0);
  v4f b = *(const v4f*)(p + 4);
  v4f c = *(const v4f*)(p + 16);
  v4f d = *(const v4f*)(p + 20);
  v16h r;
  r[0]=(_Float16)a.x;  r[1]=(_Float16)a.y;  r[2]=(_Float16)a.z;  r[3]=(_Float16)a.w;
  r[4]=(_Float16)b.x;  r[5]=(_Float16)b.y;  r[6]=(_Float16)b.z;  r[7]=(_Float16)b.w;
  r[8]=(_Float16)c.x;  r[9]=(_Float16)c.y;  r[10]=(_Float16)c.z; r[11]=(_Float16)c.w;
  r[12]=(_Float16)d.x; r[13]=(_Float16)d.y; r[14]=(_Float16)d.z; r[15]=(_Float16)d.w;
  return r;
}

// 16 f16 from LDS: 8 at base, 8 at base+hi_off (two ds_load_b128).
static __device__ inline v16h lds_frag(const _Float16* base, int hi_off) {
  v8h lo = *(const v8h*)(base);
  v8h hi = *(const v8h*)(base + hi_off);
  return __builtin_shufflevector(lo, hi, 0,1,2,3,4,5,6,7,8,9,10,11,12,13,14,15);
}

__global__ void __launch_bounds__(NTHR)
fattn_gfx1250(const float* __restrict__ Q, const float* __restrict__ K,
              const float* __restrict__ V, float* __restrict__ Out) {
  extern __shared__ unsigned char smem[];
  _Float16* Klds = (_Float16*)(smem + OFF_K);
  _Float16* Vt   = (_Float16*)(smem + OFF_V);
  float*    Sred = (float*)   (smem + OFF_S);
  _Float16* Plds = (_Float16*)(smem + OFF_P);

  const int tid    = threadIdx.x;
  const int wv     = tid >> 5;        // wave 0..15
  const int g      = wv >> 3;         // M-tile group 0/1
  const int wq     = wv & 7;          // K-slice / D-slice index within group
  const int lane   = tid & 31;
  const int lm     = lane & 15;       // position within 16-lane half
  const int lh     = lane >> 4;       // 0: lanes 0-15, 1: lanes 16-31
  const int m_base = blockIdx.x * BM + g * 16;

  // ---- Q A-fragments, resident in VGPRs for the whole kernel ----
  v16h qf[4];
  {
    const float* qrow =
        Q + (size_t)(m_base + lm) * DMODEL + wq * KSPAN + lh * 8;
#pragma unroll
    for (int kc = 0; kc < 4; ++kc) qf[kc] = a_frag_f32(qrow + kc * 32);
  }

  v8f Oc[8];
#pragma unroll
  for (int c = 0; c < 8; ++c) Oc[c] = (v8f){};
  float mrun[8], lrun[8];
#pragma unroll
  for (int r = 0; r < 8; ++r) { mrun[r] = -1e30f; lrun[r] = 0.0f; }

  // causal: both 16-row tiles of this block need j blocks 0..blockIdx.x
  const int jb_end = blockIdx.x;

  // LDS bases for this wave's B fragments (lane-dependent, loop-invariant)
  const _Float16* kb0 = &Klds[(0 * 16 + lm) * DMODEL + wq * KSPAN + lh * 8];
  const _Float16* kb1 = &Klds[(1 * 16 + lm) * DMODEL + wq * KSPAN + lh * 8];
  const _Float16* vtb = &Vt[(wq * KSPAN + lm) * BN + lh * 8];

  for (int jb = 0; jb <= jb_end; ++jb) {
    __syncthreads();   // previous iteration's LDS consumers are done

    // ---- stage K_j row-major f16 and V_j transposed f16 into LDS ----
    {
      const v4f* Kg = (const v4f*)(K + (size_t)jb * BN * DMODEL);
      const v4f* Vg = (const v4f*)(V + (size_t)jb * BN * DMODEL);
#pragma unroll
      for (int i = 0; i < 16; ++i) {   // 8192 float4s over 512 threads
        int idx4 = tid + i * NTHR;
        int row  = idx4 >> 8;          // / (1024/4)
        int col  = (idx4 & 255) << 2;
        v4f kk = Kg[idx4];
        v4h kp;
        kp.x = (_Float16)kk.x; kp.y = (_Float16)kk.y;
        kp.z = (_Float16)kk.z; kp.w = (_Float16)kk.w;
        *(v4h*)&Klds[row * DMODEL + col] = kp;      // ds_store_b64
      }
      // V: row-pairs so transposed f16 pairs pack into one ds_store_b32
#pragma unroll
      for (int i = 0; i < 8; ++i) {    // 16 rowpairs * 256 col4s over 512 threads
        int u   = tid + i * NTHR;
        int p2  = u >> 8;              // rowpair 0..15
        int c4  = u & 255;
        int col = c4 << 2;
        v4f a = Vg[(p2 * 2 + 0) * 256 + c4];
        v4f b = Vg[(p2 * 2 + 1) * 256 + c4];
#pragma unroll
        for (int e = 0; e < 4; ++e) {
          v2h pr;
          pr.x = (_Float16)a[e];
          pr.y = (_Float16)b[e];
          *(v2h*)&Vt[(col + e) * BN + p2 * 2] = pr; // ds_store_b32
        }
      }
    }
    if (jb < jb_end) {   // L2 prefetch hint for next tile (uniform branch)
      __builtin_prefetch((const char*)(K + (size_t)(jb + 1) * BN * DMODEL) + tid * 256, 0, 1);
      __builtin_prefetch((const char*)(V + (size_t)(jb + 1) * BN * DMODEL) + tid * 256, 0, 1);
    }
    __syncthreads();

    // ---- partial S = Q[:, slice] * K_j[:, slice]^T  (16x32, k-depth 128) ----
    v8f s0 = (v8f){}, s1 = (v8f){};
#pragma unroll
    for (int kc = 0; kc < 4; ++kc) {
      v16h b0 = lds_frag(kb0 + kc * 32, 16);
      v16h b1 = lds_frag(kb1 + kc * 32, 16);
      s0 = wmma16(qf[kc], b0, s0);
      s1 = wmma16(qf[kc], b1, s1);
    }
    // Pin schedule: keep one kc of B fragments in flight over each WMMA pair.
    // This region (between the two barriers) contains exactly these 16 DS
    // reads, 8 WMMAs and 8 DS writes, so the groups bind unambiguously.
    SGB(0x100, 8, 0);   // kc0+kc1 fragment loads
    SGB(0x008, 2, 0);   // kc0 WMMAs
    SGB(0x100, 4, 0);   // kc2 loads
    SGB(0x008, 2, 0);   // kc1 WMMAs
    SGB(0x100, 4, 0);   // kc3 loads
    SGB(0x008, 2, 0);   // kc2 WMMAs
    SGB(0x008, 2, 0);   // kc3 WMMAs
    SGB(0x200, 8, 0);   // Sred partial stores

    // ---- reduce the 8 partial S tiles across each group's waves via LDS ----
#pragma unroll
    for (int r = 0; r < 8; ++r) {
      Sred[(((g * 2 + 0) * 8 + r) * 32 + lane) * 8 + wq] = s0[r];
      Sred[(((g * 2 + 1) * 8 + r) * 32 + lane) * 8 + wq] = s1[r];
    }
    __syncthreads();

    float sf[2][8];
#pragma unroll
    for (int nt = 0; nt < 2; ++nt) {
#pragma unroll
      for (int r = 0; r < 8; ++r) {
        const v4f* p = (const v4f*)&Sred[(((g * 2 + nt) * 8 + r) * 32 + lane) * 8];
        v4f x = p[0], y = p[1];
        float s = ((x.x + x.y) + (x.z + x.w)) + ((y.x + y.y) + (y.z + y.w));
        s *= (1.0f / 1024.0f);                      // reference scale = 1/d_model
        int row = m_base + r + lh * 8;              // C-layout: VGPR r -> rows r, r+8
        int col = jb * BN + nt * 16 + lm;
        sf[nt][r] = (col > row) ? -1e30f : s;       // strict causal mask
      }
    }

    // ---- online softmax: running max/sum, rescale O, emit P (f16) ----
#pragma unroll
    for (int r = 0; r < 8; ++r) {
      float mx = fmaxf(sf[0][r], sf[1][r]);
#pragma unroll
      for (int d = 8; d >= 1; d >>= 1) mx = fmaxf(mx, __shfl_xor(mx, d));
      float mnew  = fmaxf(mrun[r], mx);
      float scale = __expf(mrun[r] - mnew);
      float p0 = __expf(sf[0][r] - mnew);
      float p1 = __expf(sf[1][r] - mnew);
      float rs = p0 + p1;
#pragma unroll
      for (int d = 8; d >= 1; d >>= 1) rs += __shfl_xor(rs, d);
      lrun[r] = lrun[r] * scale + rs;
      mrun[r] = mnew;
#pragma unroll
      for (int c = 0; c < 8; ++c) Oc[c][r] = Oc[c][r] * scale;
      int rr = r + lh * 8;
      Plds[wv * 512 + rr * 32 + lm]      = (_Float16)p0;
      Plds[wv * 512 + rr * 32 + 16 + lm] = (_Float16)p1;
    }

    // wave-local LDS turnaround (Plds[wv] is private to this wave); also a
    // scheduling fence: pre-fence DS work cannot leak into the PV groups below.
    asm volatile("s_wait_dscnt 0" ::: "memory");

    // P as A-fragment (16x32): lanes 0-15 take j {0..7,16..23}, 16-31 take {8..15,24..31}
    v16h pf = lds_frag(&Plds[wv * 512 + lm * 32 + lh * 8], 16);

    // ---- O[:, wave slice] += P * V_j ----
#pragma unroll
    for (int c = 0; c < 8; ++c) {
      v16h vf = lds_frag(vtb + c * 16 * BN, 16);
      Oc[c] = wmma16(pf, vf, Oc[c]);
    }
    // Pin schedule: pf + two V fragments ahead, then 2 loads per WMMA.
    SGB(0x100, 6, 1);   // pf (2) + V c0,c1 (4)
    SGB(0x008, 1, 1);   // c0 WMMA
    SGB(0x100, 2, 1);  SGB(0x008, 1, 1);   // c2 loads / c1 WMMA
    SGB(0x100, 2, 1);  SGB(0x008, 1, 1);
    SGB(0x100, 2, 1);  SGB(0x008, 1, 1);
    SGB(0x100, 2, 1);  SGB(0x008, 1, 1);
    SGB(0x100, 2, 1);  SGB(0x008, 1, 1);
    SGB(0x100, 2, 1);  SGB(0x008, 1, 1);
    SGB(0x008, 1, 1);   // c7 WMMA
  }

  // ---- epilogue: O / l, fp32 out ----
#pragma unroll
  for (int r = 0; r < 8; ++r) {
    float inv = 1.0f / lrun[r];
    int row = m_base + r + lh * 8;
    float* orow = Out + (size_t)row * DMODEL + wq * KSPAN + lm;
#pragma unroll
    for (int c = 0; c < 8; ++c) orow[c * 16] = Oc[c][r] * inv;
  }
}

extern "C" void kernel_launch(void* const* d_in, const int* in_sizes, int n_in,
                              void* d_out, int out_size, void* d_ws, size_t ws_size,
                              hipStream_t stream) {
  const float* Q = (const float*)d_in[0];
  const float* K = (const float*)d_in[1];
  const float* V = (const float*)d_in[2];
  float* out = (float*)d_out;
  (void)in_sizes; (void)n_in; (void)out_size; (void)d_ws; (void)ws_size;

  // Idempotent host-side attribute (needed for >64KB dynamic LDS); no guards.
  hipFuncSetAttribute((const void*)fattn_gfx1250,
                      hipFuncAttributeMaxDynamicSharedMemorySize, SMEM_BYTES);

  dim3 grid(NSEQ / BM);   // 256 workgroups
  fattn_gfx1250<<<grid, NTHR, SMEM_BYTES, stream>>>(Q, K, V, out);
}
